// _Affinity_47476568490177
// MI455X (gfx1250) — matrix operand order repaired and checked by reading, not compiled, not run on hardware
//
#include <hip/hip_runtime.h>
#include <hip/hip_bf16.h>

typedef _Float16 f16;
typedef __attribute__((ext_vector_type(16))) _Float16 v16h;
typedef __attribute__((ext_vector_type(8)))  _Float16 v8h;
typedef __attribute__((ext_vector_type(8)))  float    v8f;

#define B_      8
#define CIN     256
#define CMID    512
#define COUT    256
#define NPTS    2048
#define SCALE_  10.0f
#define EPS_    1e-3f
#define SLOPE_  0.01f
#define NORMEPS 1e-12f

// ---- WMMA fragment helpers (wave32 VGPR layouts, CDNA5 ISA) ----

// A-matrix 16x32 f16: lane l holds row M=l&15; K elements {b8..b8+7, b8+16..b8+23},
// b8=(l>>4)*8.  p points at (row_base + k0 + b8); two 16-byte loads.
__device__ __forceinline__ v16h frag_a(const f16* p) {
  v8h lo = *(const v8h*)(p);
  v8h hi = *(const v8h*)(p + 16);
  v16h r;
#pragma unroll
  for (int i = 0; i < 8; ++i) { r[i] = lo[i]; r[i + 8] = hi[i]; }
  return r;
}

// B-matrix 32x16 f16: lane l holds column N=l&15; K elements {kc..kc+7, kc+16..kc+23},
// kc=(l>>4)*8.  p points at (col_base + k0 + kc); two 16-byte loads.
__device__ __forceinline__ v16h frag_b(const f16* p) {
  v8h lo = *(const v8h*)(p);
  v8h hi = *(const v8h*)(p + 16);
  v16h r;
#pragma unroll
  for (int i = 0; i < 8; ++i) { r[i] = lo[i]; r[i + 8] = hi[i]; }
  return r;
}

__device__ __forceinline__ v8f wmma_f16(v16h a, v16h b, v8f c) {
  return __builtin_amdgcn_wmma_f32_16x16x32_f16(
      /*neg_a=*/false, a, /*neg_b=*/false, b,
      /*c_mod=*/(short)0, c, /*reuse_a=*/false, /*reuse_b=*/false);
}

// Async global->LDS copy of one 16-byte chunk (gfx1250 VGLOBAL async path,
// tracked by ASYNCcnt; CDNA5 ISA).  GV addressing:
// VDST = 32-bit LDS byte address, VADDR = 64-bit global byte address.
__device__ __forceinline__ void async_copy_b128(void* lds_dst, const void* gsrc) {
  unsigned lds_addr = (unsigned)(unsigned long long)lds_dst;  // low 32b = LDS offset
  unsigned long long gaddr = (unsigned long long)gsrc;
  asm volatile("global_load_async_to_lds_b128 %0, %1, off"
               :: "v"(lds_addr), "v"(gaddr) : "memory");
}
__device__ __forceinline__ void wait_asynccnt0() {
  asm volatile("s_wait_asynccnt 0x0" ::: "memory");
}

// ---- Kernel 1: fp32 -> f16 weight conversion ----
__global__ __launch_bounds__(256) void convert_weights(
    const float* __restrict__ W1, const float* __restrict__ W2,
    f16* __restrict__ W1h, f16* __restrict__ W2h) {
  int i = blockIdx.x * 256 + threadIdx.x;           // 0..262143
  if (i < CMID * CIN) W1h[i] = (f16)W1[i];
  else                W2h[i - CMID * CIN] = (f16)W2[i - CMID * CIN];
}

// ---- Kernel 2: y = l2norm(W2 * leaky(W1 * x)), output f16 [b][n][c] ----
__global__ __launch_bounds__(256) void feature_kernel(
    const float* __restrict__ x, const f16* __restrict__ W1h,
    const f16* __restrict__ W2h, f16* __restrict__ featT) {
  __shared__ f16   xT[16][CIN];     // 8 KB  : x tile, [col][k]
  __shared__ f16   hT[16][CMID];    // 16 KB : hidden,  [col][k]
  __shared__ float yT[16][COUT];    // 16 KB : output,  [col][m]
  __shared__ float sumsq[16];

  const int blk   = blockIdx.x;     // b*128 + ntile
  const int b     = blk >> 7;
  const int ntile = blk & 127;
  const int t     = threadIdx.x;
  const int lane  = t & 31;
  const int wave  = t >> 5;
  const int lrow  = lane & 15;
  const int lhalf = lane >> 4;
  const int base8 = lhalf * 8;
  const int kchnk = lhalf * 8;

  // Stage x tile transposed: thread t owns channel c=t, 16 contiguous columns.
  {
    const float* src = x + ((size_t)b * CIN + t) * NPTS + ntile * 16;
#pragma unroll
    for (int n = 0; n < 16; ++n) xT[n][t] = (f16)src[n];
  }
  __syncthreads();

  // h = leaky(W1 * x):  M=512 -> 32 tiles, 8 waves * 4 tiles.
#pragma unroll
  for (int mt = 0; mt < 4; ++mt) {
    const int mtile = wave * 4 + mt;
    const int m = mtile * 16 + lrow;
    v8f acc = {};
#pragma unroll
    for (int k0 = 0; k0 < CIN; k0 += 32) {
      v16h a  = frag_a(W1h + (size_t)m * CIN + k0 + base8);
      v16h bb = frag_b(&xT[lrow][k0 + kchnk]);
      acc = wmma_f16(a, bb, acc);
    }
#pragma unroll
    for (int i = 0; i < 8; ++i) {
      float v = acc[i];
      v = (v >= 0.f) ? v : SLOPE_ * v;
      hT[lrow][mtile * 16 + lhalf * 8 + i] = (f16)v;   // D: N=lrow, M=lhalf*8+i
    }
  }
  __syncthreads();

  // y = W2 * h:  M=256 -> 16 tiles, 8 waves * 2 tiles, K=512.
#pragma unroll
  for (int mt = 0; mt < 2; ++mt) {
    const int mtile = wave * 2 + mt;
    const int m = mtile * 16 + lrow;
    v8f acc = {};
#pragma unroll
    for (int k0 = 0; k0 < CMID; k0 += 32) {
      v16h a  = frag_a(W2h + (size_t)m * CMID + k0 + base8);
      v16h bb = frag_b(&hT[lrow][k0 + kchnk]);
      acc = wmma_f16(a, bb, acc);
    }
#pragma unroll
    for (int i = 0; i < 8; ++i)
      yT[lrow][mtile * 16 + lhalf * 8 + i] = acc[i];
  }
  if (t < 16) sumsq[t] = 0.f;
  __syncthreads();

  // Per-column sum of squares over 256 channels (16 threads per column).
  {
    const int col = t & 15, seg = t >> 4;
    float s = 0.f;
#pragma unroll
    for (int i = 0; i < 16; ++i) { float v = yT[col][seg * 16 + i]; s += v * v; }
    atomicAdd(&sumsq[col], s);
  }
  __syncthreads();

  // Normalize and emit f16 features in [b][n][c] layout (contiguous over c).
  {
    const int col = t & 15, seg = t >> 4;
    const float rn = 1.f / fmaxf(sqrtf(sumsq[col]), NORMEPS);
    f16* dst = featT + ((size_t)b * NPTS + ntile * 16 + col) * COUT + seg * 16;
#pragma unroll
    for (int i = 0; i < 16; ++i) dst[i] = (f16)(yT[col][seg * 16 + i] * rn);
  }
}

// ---- Kernel 3: scores[b][q][k] = SCALE * <q_feat[b][q], k_feat[b][k]> ----
__global__ __launch_bounds__(256) void scores_kernel(
    const f16* __restrict__ qT, const f16* __restrict__ kT,
    float* __restrict__ out) {
  __shared__ f16 qa[16 * COUT];           // 8 KB: q-tile, [q][c], shared by all waves

  const int idx    = blockIdx.x;          // b*1024 + qtile*8 + kblock
  const int kblock = idx & 7;             // 256-wide column block
  const int qtile  = (idx >> 3) & 127;
  const int b      = idx >> 10;
  const int t      = threadIdx.x;
  const int lane   = t & 31;
  const int wave   = t >> 5;
  const int lrow   = lane & 15;
  const int lhalf  = lane >> 4;
  const int base8  = lhalf * 8;
  const int kchnk  = lhalf * 8;

  // Stage the 8 KB q-tile into LDS via the async global->LDS path:
  // 512 x 16B chunks, 2 per thread; drain ASYNCcnt, then block barrier.
  {
    const f16* src = qT + ((size_t)b * NPTS + qtile * 16) * COUT;
#pragma unroll
    for (int it = 0; it < 2; ++it) {
      const int e = (t + it * 256) * 8;   // f16 index of 16B chunk
      async_copy_b128(&qa[e], src + e);
    }
    wait_asynccnt0();
  }
  __syncthreads();

  const f16* arow = &qa[lrow * COUT];

#pragma unroll
  for (int w = 0; w < 2; ++w) {
    const int kt = kblock * 16 + wave * 2 + w;      // 16-col k-tile
    const f16* brow = kT + ((size_t)b * NPTS + kt * 16 + lrow) * COUT;
    if (w == 0) {  // prefetch next k-tile's row stream (global_prefetch_b8)
      const f16* nrow = kT + ((size_t)b * NPTS + (kt + 1) * 16 + lrow) * COUT;
      __builtin_prefetch(nrow, 0, 3);
    }
    v8f acc = {};
#pragma unroll
    for (int k0 = 0; k0 < COUT; k0 += 32) {
      v16h a  = frag_a(arow + k0 + base8);
      v16h bb = frag_b(brow + k0 + kchnk);
      acc = wmma_f16(a, bb, acc);
    }
    float* orow = out + ((size_t)b * NPTS + qtile * 16 + lhalf * 8) * NPTS
                      + (size_t)kt * 16 + lrow;
#pragma unroll
    for (int i = 0; i < 8; ++i)
      orow[(size_t)i * NPTS] = acc[i] * SCALE_;
  }
}

// ---- Kernel 4: in-place row softmax + threshold ----
__global__ __launch_bounds__(256) void softmax_kernel(float* __restrict__ out) {
  __shared__ float red[256];
  const size_t row = blockIdx.x;          // b*2048 + q
  float* p = out + row * NPTS;
  const int t = threadIdx.x;

  float vals[8];
  float m = -3.402823466e38f;
#pragma unroll
  for (int j = 0; j < 8; ++j) { vals[j] = p[t + 256 * j]; m = fmaxf(m, vals[j]); }
  red[t] = m; __syncthreads();
  for (int s = 128; s > 0; s >>= 1) {
    if (t < s) red[t] = fmaxf(red[t], red[t + s]);
    __syncthreads();
  }
  m = red[0]; __syncthreads();

  float sum = 0.f;
#pragma unroll
  for (int j = 0; j < 8; ++j) { vals[j] = __expf(vals[j] - m); sum += vals[j]; }
  red[t] = sum; __syncthreads();
  for (int s = 128; s > 0; s >>= 1) {
    if (t < s) red[t] += red[t + s];
    __syncthreads();
  }
  const float inv = 1.f / red[0];
#pragma unroll
  for (int j = 0; j < 8; ++j) {
    float w = vals[j] * inv;
    p[t + 256 * j] = (w > EPS_) ? w : 0.f;
  }
}

// ---- Kernel 5: pass key through to tail of d_out ----
__global__ __launch_bounds__(256) void copy_key(const float* __restrict__ key,
                                                float* __restrict__ dst) {
  size_t i = (size_t)blockIdx.x * 256 + threadIdx.x;
  dst[i] = key[i];
}

extern "C" void kernel_launch(void* const* d_in, const int* in_sizes, int n_in,
                              void* d_out, int out_size, void* d_ws, size_t ws_size,
                              hipStream_t stream) {
  const float* query = (const float*)d_in[0];
  const float* key   = (const float*)d_in[1];
  const float* W1    = (const float*)d_in[2];
  const float* W2    = (const float*)d_in[3];
  float* out = (float*)d_out;

  f16* W1h = (f16*)d_ws;                          // 512*256
  f16* W2h = W1h + (size_t)CMID * CIN;            // 256*512
  f16* qT  = W2h + (size_t)COUT * CMID;           // 8*2048*256
  f16* kT  = qT  + (size_t)B_ * NPTS * COUT;      // 8*2048*256
  // total ws use: ~16.5 MB of f16

  convert_weights<<<(CMID * CIN + COUT * CMID) / 256, 256, 0, stream>>>(W1, W2, W1h, W2h);
  feature_kernel<<<B_ * (NPTS / 16), 256, 0, stream>>>(query, W1h, W2h, qT);
  feature_kernel<<<B_ * (NPTS / 16), 256, 0, stream>>>(key,   W1h, W2h, kT);
  scores_kernel<<<B_ * (NPTS / 16) * (NPTS / 256), 256, 0, stream>>>(qT, kT, out);
  softmax_kernel<<<B_ * NPTS, 256, 0, stream>>>(out);
  copy_key<<<(B_ * CIN * NPTS) / 256, 256, 0, stream>>>(key, out + (size_t)B_ * NPTS * NPTS);
}
